// QuantLinearORT_22170621182416
// MI455X (gfx1250) — matrix operand order repaired and checked by reading, not compiled, not run on hardware
//
#include <hip/hip_runtime.h>

// ---- CDNA5 WMMA vector types (wave32) ----
typedef __attribute__((ext_vector_type(16))) __bf16 v16bf;
typedef __attribute__((ext_vector_type(8)))  float  v8f;

#define BM 128
#define BN 128
#define BK 64
#define LDS_STRIDE (BK + 8) /* halves; 144B row stride, 16B-aligned rows */

// Pack two f32 -> packed bf16x2 in one v_cvt_pk_bf16_f32 (RNE).
// No clang builtin is declared for it on this toolchain, but the assembler
// knows the opcode (it already appears in scalar codegen), so use inline asm.
__device__ __forceinline__ unsigned pk_bf16(float a, float b) {
    unsigned r;
    asm("v_cvt_pk_bf16_f32 %0, %1, %2" : "=v"(r) : "v"(a), "v"(b));
    return r;
}

__device__ __forceinline__ v16bf load_frag16(const __bf16* p0, const __bf16* p1) {
    union { uint4 u[2]; v16bf v; } t;
    t.u[0] = *(const uint4*)p0;   // ds_load_b128
    t.u[1] = *(const uint4*)p1;   // ds_load_b128
    return t.v;
}

// out[M,N] = x[M,K] @ dequant(W)[N,K]^T + bias
// qweight: [N][K/128][64] int32, one packed byte (2 nibbles) per element
// scales : [N * K/128] f32 ; qzeros: [N*K/128/2] int32 packed nibbles
__global__ __launch_bounds__(256)
void qlinear_wmma_bf16(const float* __restrict__ x,
                       const int*   __restrict__ qweight,
                       const float* __restrict__ scales,
                       const int*   __restrict__ qzeros,
                       const float* __restrict__ bias,
                       float* __restrict__ out,
                       int M, int N, int K)
{
    __shared__ __bf16 lA[BM * LDS_STRIDE];   // x tile, bf16        (18.0 KB)
    __shared__ __bf16 lB[BN * LDS_STRIDE];   // dequant W tile bf16 (18.0 KB)

    const int t    = threadIdx.x;
    const int wave = t >> 5;
    const int lane = t & 31;
    const int lrow = lane & 15;
    const int lhi  = lane >> 4;

    const int m0 = blockIdx.y * BM;
    const int n0 = blockIdx.x * BN;

    // wave grid: 2 (M) x 4 (N); each wave: 64x32 = 4x2 tiles of 16x16
    const int wm = (wave >> 2) * 64;
    const int wn = (wave & 3) * 32;

    v8f acc[4][2];
    #pragma unroll
    for (int i = 0; i < 4; ++i)
        #pragma unroll
        for (int j = 0; j < 2; ++j)
            acc[i][j] = (v8f){0.f, 0.f, 0.f, 0.f, 0.f, 0.f, 0.f, 0.f};

    const int nGroups = K >> 7;          // group size 128
    const int nKT     = K / BK;

    // ---- per-thread fixed addresses ------------------------------------
    // x loader: thread covers rows (t>>4)+16*i (i=0..7), cols (t&15)*4..+3
    const int xrow = t >> 4;
    const int xseg = t & 15;
    const float* xbase = x + (size_t)(m0 + xrow) * K + (xseg << 2);
    unsigned* ldsA[8];
    #pragma unroll
    for (int i = 0; i < 8; ++i)
        ldsA[i] = (unsigned*)&lA[(xrow + 16 * i) * LDS_STRIDE + (xseg << 2)];

    // W loader: thread covers n-row (t>>1), k-half (t&1)*32..+31
    const int wrow = t >> 1;
    const int whal = t & 1;
    // flat qweight offset = n*2048 + kt*32 + half*16  (linear in kt)
    const int* qbase = qweight + (size_t)(n0 + wrow) * (nGroups * 64) + (whal << 4);
    const int  sbase = (n0 + wrow) * nGroups;   // + (kt>>1)
    unsigned* ldsB = (unsigned*)&lB[wrow * LDS_STRIDE + (whal << 5)];

    // ---- staging registers (software pipeline, depth 2) ----------------
    float4 xr[8];
    int4   qr[4];
    float  sreg, nzs;   // scale, -zero*scale

    auto stage_load = [&](int kt) {
        const float* xp = xbase + (size_t)kt * BK;
        #pragma unroll
        for (int i = 0; i < 8; ++i)
            xr[i] = *(const float4*)(xp + (size_t)i * 16 * K);
        const int* qp = qbase + kt * 32;
        #pragma unroll
        for (int j = 0; j < 4; ++j)
            qr[j] = *(const int4*)(qp + (j << 2));
        const int sidx = sbase + (kt >> 1);
        const int zb   = qzeros[sidx >> 1];
        const float z  = (float)((sidx & 1) ? ((zb >> 4) & 15) : (zb & 15));
        sreg = scales[sidx];
        nzs  = -z * sreg;
    };

    auto stage_commit = [&]() {
        #pragma unroll
        for (int i = 0; i < 8; ++i) {
            ldsA[i][0] = pk_bf16(xr[i].x, xr[i].y);
            ldsA[i][1] = pk_bf16(xr[i].z, xr[i].w);
        }
        const float s = sreg, c = nzs;   // w = q*s + (-z*s)
        #pragma unroll
        for (int j = 0; j < 4; ++j) {
            const int4 q4 = qr[j];
            ldsB[4 * j + 0] = pk_bf16(fmaf((float)(q4.x & 15), s, c),
                                      fmaf((float)((q4.x >> 4) & 15), s, c));
            ldsB[4 * j + 1] = pk_bf16(fmaf((float)(q4.y & 15), s, c),
                                      fmaf((float)((q4.y >> 4) & 15), s, c));
            ldsB[4 * j + 2] = pk_bf16(fmaf((float)(q4.z & 15), s, c),
                                      fmaf((float)((q4.z >> 4) & 15), s, c));
            ldsB[4 * j + 3] = pk_bf16(fmaf((float)(q4.w & 15), s, c),
                                      fmaf((float)((q4.w >> 4) & 15), s, c));
        }
    };

    // ---- main K loop ----------------------------------------------------
    stage_load(0);
    for (int kt = 0; kt < nKT; ++kt) {
        stage_commit();                 // regs (tile kt) -> LDS
        __syncthreads();
        if (kt + 1 < nKT)
            stage_load(kt + 1);         // issue next tile's global loads now

        #pragma unroll
        for (int kk = 0; kk < BK; kk += 32) {
            // preload all fragments of this k-slice, then burst 8 WMMAs
            v16bf bfr[2], afr[4];
            #pragma unroll
            for (int j = 0; j < 2; ++j) {
                // B: lane = column n, contiguous 16 K per lane-half
                const __bf16* p = &lB[(wn + j * 16 + lrow) * LDS_STRIDE + kk + (lhi << 4)];
                bfr[j] = load_frag16(p, p + 8);
            }
            #pragma unroll
            for (int i = 0; i < 4; ++i) {
                // A: lane = row m, contiguous 16 K per lane-half
                const __bf16* p = &lA[(wm + i * 16 + lrow) * LDS_STRIDE + kk + (lhi << 4)];
                afr[i] = load_frag16(p, p + 8);
            }
            #pragma unroll
            for (int i = 0; i < 4; ++i)
                #pragma unroll
                for (int j = 0; j < 2; ++j)
                    acc[i][j] = __builtin_amdgcn_wmma_f32_16x16x32_bf16(
                        false, afr[i], false, bfr[j],
                        (short)0, acc[i][j], false, false);
        }
        __syncthreads();
    }

    // ---- epilogue: D layout -> global, add bias -------------------------
    // VGPR r: M = r + 8*(lane>=16), N = lane%16
    #pragma unroll
    for (int i = 0; i < 4; ++i) {
        const int mbase = m0 + wm + i * 16 + (lhi << 3);
        #pragma unroll
        for (int j = 0; j < 2; ++j) {
            const int col  = n0 + wn + j * 16 + lrow;
            const float bb = bias[col];
            #pragma unroll
            for (int r = 0; r < 8; ++r) {
                out[(size_t)(mbase + r) * N + col] = acc[i][j][r] + bb;
            }
        }
    }
}

extern "C" void kernel_launch(void* const* d_in, const int* in_sizes, int n_in,
                              void* d_out, int out_size, void* d_ws, size_t ws_size,
                              hipStream_t stream) {
    const float* x       = (const float*)d_in[0];
    const int*   qweight = (const int*)  d_in[1];
    const float* scales  = (const float*)d_in[2];
    const int*   qzeros  = (const int*)  d_in[3];
    const float* bias    = (const float*)d_in[4];
    float*       out     = (float*)d_out;

    const int K = 4096;
    const int N = 4096;
    const int M = in_sizes[0] / K;   // 4*2048 = 8192

    dim3 grid(N / BN, M / BM);       // 32 x 64 blocks
    qlinear_wmma_bf16<<<grid, 256, 0, stream>>>(x, qweight, scales, qzeros, bias, out, M, N, K);
    (void)d_ws; (void)ws_size; (void)n_in; (void)out_size;
}
